// E2E_PextE_76312978916103
// MI455X (gfx1250) — matrix operand order repaired and checked by reading, not compiled, not run on hardware
//
#include <hip/hip_runtime.h>
#include <hip/hip_bf16.h>

#define EMB 200
#define EMB_POS 50
#define SEN 30
#define DOC 75
#define HH 100
#define NC 2
#define BB 32
#define H2 200      // 2H
#define G4 400      // 4H gate width
#define KX 224      // padded LSTM input K (>= 202), multiple of 32
#define KH 128      // padded hidden K, multiple of 32
#define KP 480      // padded pair K (450)
#define NP 64       // padded pair N (50)

typedef __attribute__((ext_vector_type(16))) _Float16 v16h;
typedef __attribute__((ext_vector_type(8)))  _Float16 v8h;
typedef __attribute__((ext_vector_type(8)))  float    v8f;

union AF { v16h v; v8h h[2]; };

// -------------------- utility kernels --------------------

// f32 [N,K] -> f16 [Np,Kp], zero padded rows/cols
__global__ void cvt_pad(const float* __restrict__ src, _Float16* __restrict__ dst,
                        int N, int K, int Np, int Kp) {
    int i = blockIdx.x * blockDim.x + threadIdx.x;
    if (i >= Np * Kp) return;
    int n = i / Kp, k = i % Kp;
    dst[i] = (n < N && k < K) ? (_Float16)src[(size_t)n * K + k] : (_Float16)0.f;
}

__global__ void vadd(const float* __restrict__ a, const float* __restrict__ b,
                     float* __restrict__ o, int n) {
    int i = blockIdx.x * blockDim.x + threadIdx.x;
    if (i < n) o[i] = a[i] + b[i];
}

// -------------------- WMMA LSTM --------------------
// X: f16 [N, T, KX] (zero padded). One block owns 16 rows, iterates T steps.
// Whh is staged once into LDS; x_t tiles are double-buffered via async LDS loads.
__global__ void lstm_wmma(const _Float16* __restrict__ X,
                          const _Float16* __restrict__ Wih,   // [G4, KX] f16
                          const _Float16* __restrict__ Whh,   // [G4, KH] f16
                          const float*    __restrict__ bias,  // [G4] = bih + bhh
                          float*          __restrict__ Hout,  // [N, T, H2]
                          int T, int outOffset, int reverse) {
    __shared__ _Float16 lA[2][16 * KX];   // double-buffered input tile (2 x 7 KB)
    __shared__ _Float16 lW[G4 * KH];      // recurrent weights (100 KB)
    __shared__ _Float16 lH[16 * KH];      // hidden f16
    __shared__ float    lG[16 * G4];      // gates
    __shared__ float    lC[16 * HH];      // cell state

    const int tid  = threadIdx.x;
    const int wave = tid >> 5;
    const int lane = tid & 31;
    const int m0   = blockIdx.x * 16;

    // stage Whh into LDS (v8h copies), init h/c
    for (int i = tid; i < (G4 * KH) / 8; i += 256)
        *(v8h*)(lW + i * 8) = *(const v8h*)(Whh + i * 8);
    for (int i = tid; i < 16 * KH; i += 256) lH[i] = (_Float16)0.f;
    for (int i = tid; i < 16 * HH; i += 256) lC[i] = 0.f;

    // async-stage one x_t tile (16 rows x 448B) into lA[buf]
    auto stage_async = [&](int ts, int buf) {
        unsigned lbase = (unsigned)(size_t)(const void*)(lA[buf]);
        #pragma unroll
        for (int r = 0; r < 2; ++r) {
            int idx = tid + r * 256;                 // 16B chunk id, 28 per row
            if (idx < 16 * (KX / 8)) {
                int row = idx / (KX / 8);
                int chunk = idx % (KX / 8);
                const void* g = (const void*)(X + ((size_t)(m0 + row) * T + ts) * KX + chunk * 8);
                unsigned l = lbase + (unsigned)(row * (KX * 2) + chunk * 16);
                asm volatile("global_load_async_to_lds_b128 %0, %1, off"
                             :: "v"(l), "v"(g) : "memory");
            }
        }
    };

    stage_async(reverse ? (T - 1) : 0, 0);

    const int mrow  = lane & 15;
    const int koff  = (lane < 16) ? 0 : 8;    // A-fragment K base
    const int kboff = (lane < 16) ? 0 : 8;    // B-fragment K base
    const int crow  = (lane < 16) ? 0 : 8;    // C/D row base
    const int ccol  = lane & 15;

    for (int t = 0; t < T; ++t) {
        const int ts = reverse ? (T - 1 - t) : t;
        asm volatile("s_wait_asynccnt 0x0" ::: "memory");
        __syncthreads();
        if (t + 1 < T) stage_async(reverse ? (T - 2 - t) : (t + 1), (t + 1) & 1);

        const _Float16* lAbuf = lA[t & 1];

        // 25 gate n-tiles over 8 waves
        for (int nt = wave; nt < G4 / 16; nt += 8) {
            const int n0 = nt * 16;
            v8f acc = {};
            const _Float16* arow = lAbuf + mrow * KX;
            const _Float16* brow = Wih + (size_t)(n0 + mrow) * KX + kboff;
            #pragma unroll
            for (int kk = 0; kk < KX / 32; ++kk) {
                const int k0 = kk * 32;
                AF a, b;
                a.h[0] = *(const v8h*)(arow + k0 + koff);
                a.h[1] = *(const v8h*)(arow + k0 + 16 + koff);
                b.h[0] = *(const v8h*)(brow + k0);
                b.h[1] = *(const v8h*)(brow + k0 + 16);
                acc = __builtin_amdgcn_wmma_f32_16x16x32_f16(
                        false, a.v, false, b.v, (short)0, acc, false, false);
            }
            const _Float16* hrow  = lH + mrow * KH;
            const _Float16* brow2 = lW + (n0 + mrow) * KH + kboff;
            #pragma unroll
            for (int kk = 0; kk < KH / 32; ++kk) {
                const int k0 = kk * 32;
                AF a, b;
                a.h[0] = *(const v8h*)(hrow + k0 + koff);
                a.h[1] = *(const v8h*)(hrow + k0 + 16 + koff);
                b.h[0] = *(const v8h*)(brow2 + k0);
                b.h[1] = *(const v8h*)(brow2 + k0 + 16);
                acc = __builtin_amdgcn_wmma_f32_16x16x32_f16(
                        false, a.v, false, b.v, (short)0, acc, false, false);
            }
            #pragma unroll
            for (int rr = 0; rr < 8; ++rr)
                lG[(crow + rr) * G4 + n0 + ccol] = acc[rr];
        }
        __syncthreads();

        // gate nonlinearity: PyTorch order i,f,g,o
        for (int i = tid; i < 16 * HH; i += 256) {
            int r = i / HH, hh = i % HH;
            float gi = lG[r * G4 + hh]          + bias[hh];
            float gf = lG[r * G4 + HH + hh]     + bias[HH + hh];
            float gg = lG[r * G4 + 2 * HH + hh] + bias[2 * HH + hh];
            float go = lG[r * G4 + 3 * HH + hh] + bias[3 * HH + hh];
            float si = 1.f / (1.f + __expf(-gi));
            float sf = 1.f / (1.f + __expf(-gf));
            float so = 1.f / (1.f + __expf(-go));
            float c  = sf * lC[i] + si * tanhf(gg);
            float hv = so * tanhf(c);
            lC[i] = c;
            lH[r * KH + hh] = (_Float16)hv;
            Hout[((size_t)(m0 + r) * T + ts) * H2 + outOffset + hh] = hv;
        }
        __syncthreads();
    }
}

// -------------------- attention pooling --------------------
__global__ void attention(const float* __restrict__ Hw,
                          const float* __restrict__ W1, const float* __restrict__ b1,
                          const float* __restrict__ W2, const float* __restrict__ b2,
                          float* __restrict__ sOut, _Float16* __restrict__ sx16) {
    __shared__ float lh[SEN * H2];
    __shared__ float lu[SEN * H2];
    __shared__ float sc[SEN];
    const int n = blockIdx.x, tid = threadIdx.x;
    const float* hrow = Hw + (size_t)n * SEN * H2;
    for (int i = tid; i < SEN * H2; i += 256) lh[i] = hrow[i];
    __syncthreads();
    for (int i = tid; i < SEN * H2; i += 256) {
        int ti = i / H2, c = i % H2;
        float acc = b1[c];
        const float* w = W1 + (size_t)c * H2;
        for (int k = 0; k < H2; ++k) acc += lh[ti * H2 + k] * w[k];
        lu[i] = tanhf(acc);
    }
    __syncthreads();
    for (int ti = tid; ti < SEN; ti += 256) {
        float acc = b2[0];
        for (int k = 0; k < H2; ++k) acc += lu[ti * H2 + k] * W2[k];
        sc[ti] = acc;
    }
    __syncthreads();
    if (tid == 0) {
        float m = sc[0];
        for (int t = 1; t < SEN; ++t) m = fmaxf(m, sc[t]);
        float ssum = 0.f;
        for (int t = 0; t < SEN; ++t) { float e = __expf(sc[t] - m); sc[t] = e; ssum += e; }
        float inv = 1.f / ssum;
        for (int t = 0; t < SEN; ++t) sc[t] *= inv;
    }
    __syncthreads();
    for (int k = tid; k < KX; k += 256) {
        if (k < H2) {
            float acc = 0.f;
            for (int t = 0; t < SEN; ++t) acc += sc[t] * lh[t * H2 + k];
            sOut[(size_t)n * H2 + k] = acc;
            sx16[(size_t)n * KX + k] = (_Float16)acc;
        } else {
            sx16[(size_t)n * KX + k] = (_Float16)0.f;
        }
    }
}

// -------------------- NC=2 projection + softmax --------------------
__global__ void proj2(const float* __restrict__ X, const float* __restrict__ W,
                      const float* __restrict__ b, float* __restrict__ out, int rows) {
    int r = blockIdx.x * blockDim.x + threadIdx.x;
    if (r >= rows) return;
    const float* x = X + (size_t)r * H2;
    float z0 = b[0], z1 = b[1];
    for (int k = 0; k < H2; ++k) { z0 += x[k] * W[k]; z1 += x[k] * W[H2 + k]; }
    float m = fmaxf(z0, z1);
    float e0 = __expf(z0 - m), e1 = __expf(z1 - m);
    float inv = 1.f / (e0 + e1);
    out[(size_t)r * 2]     = e0 * inv;
    out[(size_t)r * 2 + 1] = e1 * inv;
}

// cause LSTM input: concat(s[200], pred_pos[2]) -> f16 [rows, KX]
__global__ void build_cx(const float* __restrict__ s, const float* __restrict__ pp,
                         _Float16* __restrict__ cx, int rows) {
    int i = blockIdx.x * blockDim.x + threadIdx.x;
    if (i >= rows * KX) return;
    int n = i / KX, k = i % KX;
    float v = (k < H2) ? s[(size_t)n * H2 + k]
            : (k < H2 + NC) ? pp[(size_t)n * NC + (k - H2)] : 0.f;
    cx[i] = (_Float16)v;
}

// -------------------- WMMA pair MLP (128 threads / 4 waves) --------------------
__global__ void pair_wmma(const float* __restrict__ xp, const float* __restrict__ xc,
                          const float* __restrict__ dist,
                          const _Float16* __restrict__ W1h,   // [NP, KP] f16 padded
                          const float* __restrict__ b1,
                          const float* __restrict__ W2, const float* __restrict__ b2,
                          float* __restrict__ out) {
    __shared__ _Float16 lF[16 * KP];
    __shared__ float    lM[16 * NP];
    const int tid = threadIdx.x, wave = tid >> 5, lane = tid & 31;
    const int r0 = blockIdx.x * 16;

    for (int i = tid; i < 16 * KP; i += 128) {
        int r = i / KP, k = i % KP;
        int gr = r0 + r;
        int bI = gr / (DOC * DOC);
        int pq = gr % (DOC * DOC);
        int p = pq / DOC, q = pq % DOC;
        float v;
        if      (k < H2)               v = xp[((size_t)bI * DOC + p) * H2 + k];
        else if (k < 2 * H2)           v = xc[((size_t)bI * DOC + q) * H2 + (k - H2)];
        else if (k < 2 * H2 + EMB_POS) v = dist[(size_t)gr * EMB_POS + (k - 2 * H2)];
        else v = 0.f;
        lF[i] = (_Float16)v;
    }
    __syncthreads();

    {
        const int n0 = wave * 16;
        const int mrow  = lane & 15;
        const int koff  = (lane < 16) ? 0 : 8;
        const int kboff = (lane < 16) ? 0 : 8;
        v8f acc = {};
        const _Float16* arow = lF + mrow * KP;
        const _Float16* brow = W1h + (size_t)(n0 + mrow) * KP + kboff;
        #pragma unroll
        for (int kk = 0; kk < KP / 32; ++kk) {
            const int k0 = kk * 32;
            AF a, b;
            a.h[0] = *(const v8h*)(arow + k0 + koff);
            a.h[1] = *(const v8h*)(arow + k0 + 16 + koff);
            b.h[0] = *(const v8h*)(brow + k0);
            b.h[1] = *(const v8h*)(brow + k0 + 16);
            acc = __builtin_amdgcn_wmma_f32_16x16x32_f16(
                    false, a.v, false, b.v, (short)0, acc, false, false);
        }
        const int crow = (lane < 16) ? 0 : 8;
        const int col  = n0 + (lane & 15);
        const float bb = (col < 50) ? b1[col] : 0.f;
        #pragma unroll
        for (int rr = 0; rr < 8; ++rr)
            lM[(crow + rr) * NP + col] = fmaxf(acc[rr] + bb, 0.f);
    }
    __syncthreads();

    if (tid < 16) {
        int gr = r0 + tid;
        float z0 = b2[0], z1 = b2[1];
        for (int k = 0; k < 50; ++k) {
            float hv = lM[tid * NP + k];
            z0 += hv * W2[k];
            z1 += hv * W2[50 + k];
        }
        float m = fmaxf(z0, z1);
        float e0 = __expf(z0 - m), e1 = __expf(z1 - m);
        float inv = 1.f / (e0 + e1);
        out[(size_t)gr * 2]     = e0 * inv;
        out[(size_t)gr * 2 + 1] = e1 * inv;
    }
}

// -------------------- host launcher --------------------
extern "C" void kernel_launch(void* const* d_in, const int* in_sizes, int n_in,
                              void* d_out, int out_size, void* d_ws, size_t ws_size,
                              hipStream_t stream) {
    (void)in_sizes; (void)n_in; (void)out_size; (void)ws_size;
    const float* x    = (const float*)d_in[0];
    const float* dist = (const float*)d_in[1];
    const float* wW[8]; const float* pW[8]; const float* cW[8];
    for (int i = 0; i < 8; ++i) { wW[i] = (const float*)d_in[2 + i];
                                  pW[i] = (const float*)d_in[10 + i];
                                  cW[i] = (const float*)d_in[18 + i]; }
    const float* att_W1 = (const float*)d_in[26];
    const float* att_b1 = (const float*)d_in[27];
    const float* att_W2 = (const float*)d_in[28];
    const float* att_b2 = (const float*)d_in[29];
    const float* pos_W  = (const float*)d_in[30];
    const float* pos_b  = (const float*)d_in[31];
    const float* cau_W  = (const float*)d_in[32];
    const float* cau_b  = (const float*)d_in[33];
    const float* pr_W1  = (const float*)d_in[34];
    const float* pr_b1  = (const float*)d_in[35];
    const float* pr_W2  = (const float*)d_in[36];
    const float* pr_b2  = (const float*)d_in[37];

    char* wsp = (char*)d_ws;
    size_t off = 0;
    auto alloc = [&](size_t bytes) -> void* {
        void* p = wsp + off;
        off = (off + bytes + 255) & ~(size_t)255;
        return p;
    };
    const int NW = BB * DOC;                 // 2400 word sequences
    _Float16* Xw16  = (_Float16*)alloc((size_t)NW * SEN * KX * 2);
    _Float16* wWih16[2] = { (_Float16*)alloc((size_t)G4 * KX * 2), (_Float16*)alloc((size_t)G4 * KX * 2) };
    _Float16* wWhh16[2] = { (_Float16*)alloc((size_t)G4 * KH * 2), (_Float16*)alloc((size_t)G4 * KH * 2) };
    _Float16* pWih16[2] = { (_Float16*)alloc((size_t)G4 * KX * 2), (_Float16*)alloc((size_t)G4 * KX * 2) };
    _Float16* pWhh16[2] = { (_Float16*)alloc((size_t)G4 * KH * 2), (_Float16*)alloc((size_t)G4 * KH * 2) };
    _Float16* cWih16[2] = { (_Float16*)alloc((size_t)G4 * KX * 2), (_Float16*)alloc((size_t)G4 * KX * 2) };
    _Float16* cWhh16[2] = { (_Float16*)alloc((size_t)G4 * KH * 2), (_Float16*)alloc((size_t)G4 * KH * 2) };
    _Float16* prW1h = (_Float16*)alloc((size_t)NP * KP * 2);
    float* wB[2] = { (float*)alloc(G4 * 4), (float*)alloc(G4 * 4) };
    float* pB[2] = { (float*)alloc(G4 * 4), (float*)alloc(G4 * 4) };
    float* cB[2] = { (float*)alloc(G4 * 4), (float*)alloc(G4 * 4) };
    float* Hw      = (float*)alloc((size_t)NW * SEN * H2 * 4);
    float* sBuf    = (float*)alloc((size_t)NW * H2 * 4);
    _Float16* sx16 = (_Float16*)alloc((size_t)NW * KX * 2);
    _Float16* cx16 = (_Float16*)alloc((size_t)NW * KX * 2);
    float* x_pos   = (float*)alloc((size_t)NW * H2 * 4);
    float* x_cause = (float*)alloc((size_t)NW * H2 * 4);

    float* pred_pos   = (float*)d_out;                // [2400,2]
    float* pred_cause = (float*)d_out + 4800;
    float* pred_pair  = (float*)d_out + 9600;         // [180000,2]

    auto cvt = [&](const float* src, _Float16* dst, int N, int K, int Np, int Kp) {
        int tot = Np * Kp;
        cvt_pad<<<(tot + 255) / 256, 256, 0, stream>>>(src, dst, N, K, Np, Kp);
    };
    cvt(x, Xw16, NW * SEN, EMB, NW * SEN, KX);
    for (int d = 0; d < 2; ++d) {
        cvt(wW[0 + 4 * d], wWih16[d], G4, EMB, G4, KX);
        cvt(wW[1 + 4 * d], wWhh16[d], G4, HH,  G4, KH);
        vadd<<<2, 256, 0, stream>>>(wW[2 + 4 * d], wW[3 + 4 * d], wB[d], G4);
        cvt(pW[0 + 4 * d], pWih16[d], G4, H2,  G4, KX);
        cvt(pW[1 + 4 * d], pWhh16[d], G4, HH,  G4, KH);
        vadd<<<2, 256, 0, stream>>>(pW[2 + 4 * d], pW[3 + 4 * d], pB[d], G4);
        cvt(cW[0 + 4 * d], cWih16[d], G4, H2 + NC, G4, KX);
        cvt(cW[1 + 4 * d], cWhh16[d], G4, HH,  G4, KH);
        vadd<<<2, 256, 0, stream>>>(cW[2 + 4 * d], cW[3 + 4 * d], cB[d], G4);
    }
    cvt(pr_W1, prW1h, 50, 450, NP, KP);

    // word BiLSTM -> Hw [2400,30,200]
    lstm_wmma<<<NW / 16, 256, 0, stream>>>(Xw16, wWih16[0], wWhh16[0], wB[0], Hw, SEN, 0, 0);
    lstm_wmma<<<NW / 16, 256, 0, stream>>>(Xw16, wWih16[1], wWhh16[1], wB[1], Hw, SEN, HH, 1);

    // attention pooling -> s [2400,200], sx16 [32*75,224]
    attention<<<NW, 256, 0, stream>>>(Hw, att_W1, att_b1, att_W2, att_b2, sBuf, sx16);

    // pos BiLSTM over [32,75,.] -> x_pos
    lstm_wmma<<<BB / 16, 256, 0, stream>>>(sx16, pWih16[0], pWhh16[0], pB[0], x_pos, DOC, 0, 0);
    lstm_wmma<<<BB / 16, 256, 0, stream>>>(sx16, pWih16[1], pWhh16[1], pB[1], x_pos, DOC, HH, 1);
    proj2<<<(NW + 255) / 256, 256, 0, stream>>>(x_pos, pos_W, pos_b, pred_pos, NW);

    // cause BiLSTM over concat(s, pred_pos)
    build_cx<<<(NW * KX + 255) / 256, 256, 0, stream>>>(sBuf, pred_pos, cx16, NW);
    lstm_wmma<<<BB / 16, 256, 0, stream>>>(cx16, cWih16[0], cWhh16[0], cB[0], x_cause, DOC, 0, 0);
    lstm_wmma<<<BB / 16, 256, 0, stream>>>(cx16, cWih16[1], cWhh16[1], cB[1], x_cause, DOC, HH, 1);
    proj2<<<(NW + 255) / 256, 256, 0, stream>>>(x_cause, cau_W, cau_b, pred_cause, NW);

    // pair MLP: 32*5625 = 180000 rows
    pair_wmma<<<(BB * DOC * DOC) / 16, 128, 0, stream>>>(x_pos, x_cause, dist,
                                                         prW1h, pr_b1, pr_W2, pr_b2, pred_pair);
}
